// DCMSL_52209622450339
// MI455X (gfx1250) — matrix operand order repaired and checked by reading, not compiled, not run on hardware
//
#include <hip/hip_runtime.h>

// ---------------------------------------------------------------------------
// 2-layer GCN for MI455X (gfx1250): bf16 WMMA GEMMs + fp32 atomic scatter-add.
// Pipeline is L2/atomic bound (~2.6 GB L2 traffic; features fit in 192MB L2),
// so messages are stored bf16 (halves gather traffic) and accumulated fp32.
// ---------------------------------------------------------------------------

typedef __attribute__((ext_vector_type(16))) __bf16 v16bf;
typedef __attribute__((ext_vector_type(8)))  float  v8f;

#define FIN 256   // input feature width for both layers (layer2 K = 256 too)

__device__ __forceinline__ float bf2f(unsigned short u) {
    union { unsigned int u; float f; } c;
    c.u = ((unsigned int)u) << 16;
    return c.f;
}

// ---------------- degree / normalization ----------------
__global__ void k_init_deg(float* deg, int n) {
    int i = blockIdx.x * blockDim.x + threadIdx.x;
    if (i < n) deg[i] = 1.0f;                       // self loop contributes 1
}
__global__ void k_deg_edges(const int* __restrict__ dst, float* deg, int e) {
    int i = blockIdx.x * blockDim.x + threadIdx.x;
    if (i < e) atomicAdd(&deg[dst[i]], 1.0f);
}
__global__ void k_rsqrt_inplace(float* d, int n) {
    int i = blockIdx.x * blockDim.x + threadIdx.x;
    if (i < n) d[i] = rsqrtf(d[i]);                 // deg >= 1 always
}

// ---------------- fp32 -> bf16 (vectorized x4) ----------------
__global__ void k_f32_to_bf16(const float* __restrict__ x, __bf16* __restrict__ y, int n4) {
    int i = blockIdx.x * blockDim.x + threadIdx.x;
    if (i >= n4) return;
    float4 v = ((const float4*)x)[i];
    union { __bf16 b[4]; uint2 u; } o;
    o.b[0] = (__bf16)v.x; o.b[1] = (__bf16)v.y;
    o.b[2] = (__bf16)v.z; o.b[3] = (__bf16)v.w;
    ((uint2*)y)[i] = o.u;
}

// ---------------- pack W (K x NOUT, row-major f32) into WMMA-B fragments ----
// Fragment for (ktile,ntile): 32 lanes x 16 bf16, lane<16 -> K=kt*32+0..15,
// lane>=16 -> K=kt*32+16..31; col N = nt*16 + (lane&15); K pairs per dword.
template<int NOUT>
__global__ void k_pack_w(const float* __restrict__ W, __bf16* __restrict__ P) {
    constexpr int NT  = NOUT / 16;
    constexpr int TOT = (FIN / 32) * NT * 32;
    int t = blockIdx.x * blockDim.x + threadIdx.x;
    if (t >= TOT) return;
    int lane  = t & 31;
    int kt    = (t >> 5) / NT;
    int kbase = kt * 32 + ((lane >> 4) << 4);
    int ncol  = ((t >> 5) % NT) * 16 + (lane & 15);
    __bf16* out = P + (size_t)t * 16;
#pragma unroll
    for (int j = 0; j < 16; ++j)
        out[j] = (__bf16)W[(size_t)(kbase + j) * NOUT + ncol];
}

// ---------------- WMMA GEMM: T = A @ W  (bf16 in, f32 acc) -------------------
// block = 128 threads = 4 waves; each wave computes a 16(M) x 64(N) strip.
// Epilogue: Tb = bf16(acc)  (message source for gather),
//           Agg = acc * dinv[row]^2  (self-loop seed; fully covers buffer).
template<int NOUT>
__launch_bounds__(128)
__global__ void k_gemm_wmma(const __bf16* __restrict__ Ab,   // M x FIN bf16
                            const __bf16* __restrict__ Bp,   // packed W
                            const float*  __restrict__ dinv,
                            __bf16*       __restrict__ Tb,   // M x NOUT bf16
                            float*        __restrict__ Agg,  // M x NOUT f32
                            int mtiles) {
    constexpr int NT = NOUT / 16;
    const int lane    = threadIdx.x & 31;
    const int wave    = threadIdx.x >> 5;
    const int mt      = blockIdx.x * 4 + wave;
    if (mt >= mtiles) return;
    const int halfsel = lane >> 4;                 // 0: low K-half, 1: high
    const int arow    = mt * 16 + (lane & 15);
    const int ntBase  = blockIdx.y * 4;            // 4 n-tiles per wave

    v8f acc0 = {0.f,0.f,0.f,0.f,0.f,0.f,0.f,0.f};
    v8f acc1 = acc0, acc2 = acc0, acc3 = acc0;

    const __bf16* aRow = Ab + (size_t)arow * FIN + (halfsel << 4);
#pragma unroll
    for (int kt = 0; kt < FIN / 32; ++kt) {
        union { uint4 u[2]; v16bf v; } af;
        af.u[0] = *(const uint4*)(aRow + kt * 32);        // K 0..7  (or 16..23)
        af.u[1] = *(const uint4*)(aRow + kt * 32 + 8);    // K 8..15 (or 24..31)

        const __bf16* bBase = Bp + ((size_t)(kt * NT + ntBase) * 32 + lane) * 16;
        union { uint4 u[2]; v16bf v; } b0, b1, b2, b3;
        b0.u[0] = *(const uint4*)(bBase +    0); b0.u[1] = *(const uint4*)(bBase +    8);
        b1.u[0] = *(const uint4*)(bBase +  512); b1.u[1] = *(const uint4*)(bBase +  520);
        b2.u[0] = *(const uint4*)(bBase + 1024); b2.u[1] = *(const uint4*)(bBase + 1032);
        b3.u[0] = *(const uint4*)(bBase + 1536); b3.u[1] = *(const uint4*)(bBase + 1544);

        acc0 = __builtin_amdgcn_wmma_f32_16x16x32_bf16(false, af.v, false, b0.v, (short)0, acc0, false, false);
        acc1 = __builtin_amdgcn_wmma_f32_16x16x32_bf16(false, af.v, false, b1.v, (short)0, acc1, false, false);
        acc2 = __builtin_amdgcn_wmma_f32_16x16x32_bf16(false, af.v, false, b2.v, (short)0, acc2, false, false);
        acc3 = __builtin_amdgcn_wmma_f32_16x16x32_bf16(false, af.v, false, b3.v, (short)0, acc3, false, false);
    }

    const int col = ntBase * 16 + (lane & 15);
#pragma unroll
    for (int r = 0; r < 8; ++r) {
        int row  = mt * 16 + (halfsel ? (r + 8) : r);
        float di = dinv[row];
        float s  = di * di;
        size_t base = (size_t)row * NOUT + col;
        float v0 = acc0[r], v1 = acc1[r], v2 = acc2[r], v3 = acc3[r];
        Tb[base +  0] = (__bf16)v0;  Agg[base +  0] = v0 * s;
        Tb[base + 16] = (__bf16)v1;  Agg[base + 16] = v1 * s;
        Tb[base + 32] = (__bf16)v2;  Agg[base + 32] = v2 * s;
        Tb[base + 48] = (__bf16)v3;  Agg[base + 48] = v3 * s;
    }
}

// ---------------- edge gather/scatter: Agg[dst] += T[src] * norm -------------
// One wave per edge; lane handles F/32 contiguous channels (one b128/b64 load).
template<int F>
__launch_bounds__(256)
__global__ void k_edge_agg(const int* __restrict__ srcI, const int* __restrict__ dstI,
                           const float* __restrict__ dinv,
                           const __bf16* __restrict__ Tb,
                           float* __restrict__ Agg, int E) {
    int e = blockIdx.x * 8 + (threadIdx.x >> 5);
    if (e >= E) return;
    const int lane = threadIdx.x & 31;
    const int s = srcI[e], d = dstI[e];
    const float nrm = dinv[s] * dinv[d];
    constexpr int PER = F / 32;
    const __bf16* rp = Tb  + (size_t)s * F + lane * PER;
    float*        op = Agg + (size_t)d * F + lane * PER;
    if constexpr (PER == 8) {
        union { uint4 u; unsigned short h[8]; } v;
        v.u = *(const uint4*)rp;
#pragma unroll
        for (int j = 0; j < 8; ++j) atomicAdd(op + j, bf2f(v.h[j]) * nrm);
    } else {
        union { uint2 u; unsigned short h[4]; } v;
        v.u = *(const uint2*)rp;
#pragma unroll
        for (int j = 0; j < 4; ++j) atomicAdd(op + j, bf2f(v.h[j]) * nrm);
    }
}

// ---------------- epilogues ----------------
__global__ void k_relu_bias_bf16(const float* __restrict__ A, const float* __restrict__ b,
                                 __bf16* __restrict__ H, size_t total) {
    size_t i = (size_t)blockIdx.x * blockDim.x + threadIdx.x;
    if (i >= total) return;
    int f = (int)(i & 255);                          // F = 256
    H[i] = (__bf16)fmaxf(A[i] + b[f], 0.0f);
}
__global__ void k_relu_bias_f32(float* __restrict__ O, const float* __restrict__ b, size_t total) {
    size_t i = (size_t)blockIdx.x * blockDim.x + threadIdx.x;
    if (i >= total) return;
    int f = (int)(i & 127);                          // F = 128
    O[i] = fmaxf(O[i] + b[f], 0.0f);
}

// ---------------------------------------------------------------------------
extern "C" void kernel_launch(void* const* d_in, const int* in_sizes, int n_in,
                              void* d_out, int out_size, void* d_ws, size_t ws_size,
                              hipStream_t stream) {
    (void)n_in; (void)out_size; (void)ws_size;
    const float* x  = (const float*)d_in[0];
    const int*   ei = (const int*)  d_in[1];
    const float* W1 = (const float*)d_in[2];
    const float* b1 = (const float*)d_in[3];
    const float* W2 = (const float*)d_in[4];
    const float* b2 = (const float*)d_in[5];

    const int N = in_sizes[0] / FIN;      // 50000
    const int E = in_sizes[1] / 2;        // 800000
    const int* srcI = ei;                 // edge_index[0]
    const int* dstI = ei + E;             // edge_index[1]

    // workspace layout (bytes): dinv @0 (0.2MB) | W1p @1MB | W2p @2MB |
    // XB/H1b bf16 @3MB (24.5MB) | TB bf16 @28MB (24.5MB) | AGG f32 @53MB (49MB)
    char* ws = (char*)d_ws;
    float*  dinv = (float*) (ws + 0);
    __bf16* W1p  = (__bf16*)(ws + (size_t)(1u  << 20));
    __bf16* W2p  = (__bf16*)(ws + (size_t)(2u  << 20));
    __bf16* XB   = (__bf16*)(ws + (size_t)(3u  << 20));
    __bf16* TB   = (__bf16*)(ws + (size_t)(28u << 20));
    float*  AGG  = (float*) (ws + (size_t)(53u << 20));
    float*  OUT  = (float*)d_out;

    // normalization: deg = 1 + indegree; dinv = rsqrt(deg)
    k_init_deg     <<<(N + 255) / 256, 256, 0, stream>>>(dinv, N);
    k_deg_edges    <<<(E + 255) / 256, 256, 0, stream>>>(dstI, dinv, E);
    k_rsqrt_inplace<<<(N + 255) / 256, 256, 0, stream>>>(dinv, N);

    // input convert + weight packing
    int n4 = (N * FIN) / 4;
    k_f32_to_bf16<<<(n4 + 255) / 256, 256, 0, stream>>>(x, XB, n4);
    k_pack_w<256><<<((FIN / 32) * 16 * 32 + 255) / 256, 256, 0, stream>>>(W1, W1p);
    k_pack_w<128><<<((FIN / 32) *  8 * 32 + 255) / 256, 256, 0, stream>>>(W2, W2p);

    const int mtiles = (N + 15) / 16;     // N is a multiple of 16 here (3125 tiles)

    // layer 1: T1 = x@W1 ; Agg = self + scatter(edges) ; H1 = relu(Agg + b1) -> bf16
    dim3 g1((mtiles + 3) / 4, 256 / 64);
    k_gemm_wmma<256><<<g1, 128, 0, stream>>>(XB, W1p, dinv, TB, AGG, mtiles);
    k_edge_agg<256><<<(E + 7) / 8, 256, 0, stream>>>(srcI, dstI, dinv, TB, AGG, E);
    size_t t1 = (size_t)N * 256;
    k_relu_bias_bf16<<<(unsigned)((t1 + 255) / 256), 256, 0, stream>>>(AGG, b1, XB, t1);

    // layer 2: T2 = H1@W2 ; out = relu(self + scatter + b2), accumulated in d_out
    dim3 g2((mtiles + 3) / 4, 128 / 64);
    k_gemm_wmma<128><<<g2, 128, 0, stream>>>(XB, W2p, dinv, TB, OUT, mtiles);
    k_edge_agg<128><<<(E + 7) / 8, 256, 0, stream>>>(srcI, dstI, dinv, TB, OUT, E);
    size_t t2 = (size_t)N * 128;
    k_relu_bias_f32<<<(unsigned)((t2 + 255) / 256), 256, 0, stream>>>(OUT, b2, t2);
}
